// GNN_cell_4277787427394
// MI455X (gfx1250) — matrix operand order repaired and checked by reading, not compiled, not run on hardware
//
#include <hip/hip_runtime.h>
#include <hip/hip_bf16.h>

typedef __attribute__((ext_vector_type(16))) __bf16 v16bf;
typedef __attribute__((ext_vector_type(8)))  float  v8f;

#define DIMC 256

// ---------- helpers: order-preserving float<->uint for atomicMax on signed floats ----------
__device__ __forceinline__ unsigned enc_ord(float f) {
  unsigned u = __float_as_uint(f);
  return (u & 0x80000000u) ? ~u : (u | 0x80000000u);
}
__device__ __forceinline__ float dec_ord(unsigned u) {
  return (u & 0x80000000u) ? __uint_as_float(u ^ 0x80000000u) : __uint_as_float(~u);
}

// ---------- generic fill ----------
__global__ void fill_u32(unsigned* __restrict__ p, unsigned v, int n) {
  int i = blockIdx.x * 256 + threadIdx.x;
  if (i < n) p[i] = v;
}

// ---------- layer-0 GEMM: (N x 3) @ (3 x 256), too thin for WMMA ----------
__global__ __launch_bounds__(256) void gemm3(const float* __restrict__ X,
                                             const float* __restrict__ W,
                                             float* __restrict__ H, int N) {
  int i = blockIdx.x * 256 + threadIdx.x;
  if (i >= N * DIMC) return;
  int n = i >> 8, c = i & 255;
  const float* xr = X + (size_t)n * 3;
  H[i] = fmaf(xr[0], W[c], fmaf(xr[1], W[DIMC + c], xr[2] * W[2 * DIMC + c]));
}

// ---------- transpose + convert W (256x256 f32) -> Wt (256x256 bf16, row n = column n of W) ----------
__global__ __launch_bounds__(256) void wt_convert(const float* __restrict__ W,
                                                  __bf16* __restrict__ Wt) {
  int i = blockIdx.x * 256 + threadIdx.x;       // i over 65536
  int k = i >> 8, n = i & 255;
  Wt[(size_t)n * DIMC + k] = (__bf16)W[i];
}

// ---------- WMMA GEMM: H(Mx256) = A(Mx256,f32) @ W(256x256) via bf16 WMMA, f32 accum ----------
// One block = 4 waves = one 16-row stripe x all 256 cols. Wave w -> cols [64w,64w+64).
__global__ __launch_bounds__(128) void gemm_h(const float* __restrict__ A,
                                              const __bf16* __restrict__ Wt,
                                              float* __restrict__ H, int M) {
  const int lane  = threadIdx.x & 31;
  const int wave  = threadIdx.x >> 5;
  const int lo    = lane & 15;
  const int hi    = lane >> 4;
  const int mtile = blockIdx.x;
  const int row   = mtile * 16 + lo;            // A row this lane fetches
  const int ncol0 = wave * 64;

  const float* arow = A + (size_t)row * DIMC;

  v8f acc[4];
#pragma unroll
  for (int t = 0; t < 4; ++t) acc[t] = (v8f){0.f,0.f,0.f,0.f,0.f,0.f,0.f,0.f};

  for (int k = 0; k < DIMC; k += 32) {
    // --- A fragment (16x32 bf16): lane holds row `lo`, K = [k+hi*16, +16) contiguous
    float4 a0 = *(const float4*)(arow + k + hi * 16);
    float4 a1 = *(const float4*)(arow + k + hi * 16 + 4);
    float4 a2 = *(const float4*)(arow + k + hi * 16 + 8);
    float4 a3 = *(const float4*)(arow + k + hi * 16 + 12);
    if (k < DIMC - 32) __builtin_prefetch(arow + k + 32 + hi * 8, 0, 3);

    v16bf af;
    af[0]  = (__bf16)a0.x; af[1]  = (__bf16)a0.y; af[2]  = (__bf16)a0.z; af[3]  = (__bf16)a0.w;
    af[4]  = (__bf16)a1.x; af[5]  = (__bf16)a1.y; af[6]  = (__bf16)a1.z; af[7]  = (__bf16)a1.w;
    af[8]  = (__bf16)a2.x; af[9]  = (__bf16)a2.y; af[10] = (__bf16)a2.z; af[11] = (__bf16)a2.w;
    af[12] = (__bf16)a3.x; af[13] = (__bf16)a3.y; af[14] = (__bf16)a3.z; af[15] = (__bf16)a3.w;

#pragma unroll
    for (int t = 0; t < 4; ++t) {
      // --- B fragment (32x16 bf16): lane holds col `ncol0+16t+lo`, K = [k+hi*16, +16) contiguous
      v16bf bf = *(const v16bf*)(Wt + (size_t)(ncol0 + 16 * t + lo) * DIMC + hi * 16 + k);
      acc[t] = __builtin_amdgcn_wmma_f32_16x16x32_bf16(
          false, af, false, bf, (short)0, acc[t], false, false);
    }
  }

  // --- C/D layout: reg r -> row hi*8+r, col = tile base + lo
#pragma unroll
  for (int t = 0; t < 4; ++t) {
    int ncol = ncol0 + 16 * t + lo;
#pragma unroll
    for (int r = 0; r < 8; ++r) {
      int mrow = mtile * 16 + hi * 8 + r;
      H[(size_t)mrow * DIMC + ncol] = acc[t][r];
    }
  }
}

// ---------- per-node attention dots: hs[n]=dot(h[n],asrc), hd[n]=dot(h[n],adst) ----------
__global__ __launch_bounds__(128) void dots_kernel(const float* __restrict__ H,
                                                   const float* __restrict__ asrc,
                                                   const float* __restrict__ adst,
                                                   float* __restrict__ hs,
                                                   float* __restrict__ hd, int N) {
  int lane = threadIdx.x & 31;
  int node = blockIdx.x * 4 + (threadIdx.x >> 5);
  if (node >= N) return;
  const float* hrow = H + (size_t)node * DIMC;
  float s = 0.f, d = 0.f;
#pragma unroll
  for (int j = 0; j < 8; ++j) {
    int c = lane + 32 * j;
    float v = hrow[c];
    s = fmaf(v, asrc[c], s);
    d = fmaf(v, adst[c], d);
  }
#pragma unroll
  for (int off = 16; off > 0; off >>= 1) {
    s += __shfl_xor(s, off, 32);
    d += __shfl_xor(d, off, 32);
  }
  if (lane == 0) { hs[node] = s; hd[node] = d; }
}

// ---------- edge pass 1: segment max of leaky_relu attention logits ----------
__global__ __launch_bounds__(256) void edge_max(const int* __restrict__ src,
                                                const int* __restrict__ dst,
                                                const float* __restrict__ hs,
                                                const float* __restrict__ hd,
                                                unsigned* __restrict__ menc, int E) {
  int e = blockIdx.x * 256 + threadIdx.x;
  if (e >= E) return;
  int s = src[e], d = dst[e];
  float v = hs[s] + hd[d];
  v = v > 0.f ? v : 0.2f * v;
  atomicMax(&menc[d], enc_ord(v));
}

// ---------- edge pass 2: ex = exp(e - m[dst]); ssum[dst] += ex ----------
__global__ __launch_bounds__(256) void edge_exp(const int* __restrict__ src,
                                                const int* __restrict__ dst,
                                                const float* __restrict__ hs,
                                                const float* __restrict__ hd,
                                                const unsigned* __restrict__ menc,
                                                float* __restrict__ exb,
                                                float* __restrict__ ssum, int E) {
  int e = blockIdx.x * 256 + threadIdx.x;
  if (e >= E) return;
  int s = src[e], d = dst[e];
  float v = hs[s] + hd[d];
  v = v > 0.f ? v : 0.2f * v;
  float ex = expf(v - dec_ord(menc[d]));
  exb[e] = ex;
  atomicAdd(&ssum[d], ex);
}

// ---------- edge pass 3: OUT[dst] += alpha * h[src]  (one block per edge, 256 cols) ----------
__global__ __launch_bounds__(256) void edge_aggr(const int* __restrict__ src,
                                                 const int* __restrict__ dst,
                                                 const float* __restrict__ exb,
                                                 const float* __restrict__ ssum,
                                                 const float* __restrict__ H,
                                                 float* __restrict__ OUT) {
  int e = blockIdx.x;
  int s = src[e], d = dst[e];                 // uniform -> scalar loads
  float alpha = exb[e] / ssum[d];
  int c = threadIdx.x;
  atomicAdd(&OUT[(size_t)d * DIMC + c], alpha * H[(size_t)s * DIMC + c]);
}

// ---------- +bias, relu, segment-max pooling into next-level nodes ----------
__global__ __launch_bounds__(256) void pool_relu_max(const float* __restrict__ OUT,
                                                     const float* __restrict__ bias,
                                                     const int* __restrict__ cluster,
                                                     float* __restrict__ XN) {
  int n = blockIdx.x, c = threadIdx.x;
  float v = OUT[(size_t)n * DIMC + c] + bias[c];
  v = v > 0.f ? v : 0.f;                       // relu -> non-negative: raw bits are ordered
  atomicMax((unsigned*)&XN[(size_t)cluster[n] * DIMC + c], __float_as_uint(v));
}

// ---------- column stats: stats[c]=sum, stats[256+c]=sumsq ----------
__global__ __launch_bounds__(256) void col_stats(const float* __restrict__ X,
                                                 float* __restrict__ stats, int N) {
  int c = threadIdx.x;
  float s = 0.f, q = 0.f;
  for (int r = blockIdx.x; r < N; r += gridDim.x) {
    float v = X[(size_t)r * DIMC + c];
    s += v;
    q = fmaf(v, v, q);
  }
  atomicAdd(&stats[c], s);
  atomicAdd(&stats[DIMC + c], q);
}

__global__ void col_finalize(float* __restrict__ stats, int N) {
  int c = threadIdx.x;
  float inv = 1.f / (float)N;
  float mean = stats[c] * inv;
  float var  = stats[DIMC + c] * inv - mean * mean;
  stats[c] = mean;
  stats[DIMC + c] = rsqrtf(var + 1e-5f);
}

__global__ __launch_bounds__(256) void col_apply(const float* __restrict__ X,
                                                 const float* __restrict__ stats,
                                                 float* __restrict__ Y, int total) {
  int i = blockIdx.x * 256 + threadIdx.x;
  if (i >= total) return;
  int c = i & 255;
  Y[i] = (X[i] - stats[c]) * stats[DIMC + c];
}

extern "C" void kernel_launch(void* const* d_in, const int* in_sizes, int n_in,
                              void* d_out, int out_size, void* d_ws, size_t ws_size,
                              hipStream_t stream) {
  (void)n_in; (void)ws_size;
  const float* x_in = (const float*)d_in[0];

  int N[4];
  N[0] = in_sizes[6];            // len(cluster0) = B*706
  N[1] = in_sizes[12];           // B*512
  N[2] = in_sizes[18];           // B*256
  N[3] = out_size / DIMC;        // B*128

  int Earr[3], maxE = 0;
  for (int i = 0; i < 3; ++i) {
    Earr[i] = in_sizes[5 + 6 * i] / 2;
    if (Earr[i] > maxE) maxE = Earr[i];
  }

  // ---- carve workspace (256B aligned regions) ----
  size_t off = 0;
  auto carve = [&](size_t b) -> void* {
    void* p = (char*)d_ws + off;
    off += (b + 255) & ~(size_t)255;
    return p;
  };
  float*    Hbuf  = (float*)carve((size_t)N[0] * DIMC * 4);  // h (reused as pooled x3)
  float*    Obuf  = (float*)carve((size_t)N[0] * DIMC * 4);  // aggregation output
  float*    Xbuf  = (float*)carve((size_t)N[1] * DIMC * 4);  // pooled x1 then x2
  float*    hs    = (float*)carve((size_t)N[0] * 4);
  float*    hd    = (float*)carve((size_t)N[0] * 4);
  unsigned* menc  = (unsigned*)carve((size_t)N[0] * 4);
  float*    ssum  = (float*)carve((size_t)N[0] * 4);
  float*    exbuf = (float*)carve((size_t)maxE * 4);
  float*    stats = (float*)carve(2 * DIMC * 4);
  __bf16*   Wt    = (__bf16*)carve((size_t)DIMC * DIMC * 2);

  const unsigned ENC_NEG_INF = 0x007FFFFFu;  // enc_ord(-inf)

  for (int i = 0; i < 3; ++i) {
    const float* Wp  = (const float*)d_in[1 + 6 * i];
    const float* asr = (const float*)d_in[2 + 6 * i];
    const float* ads = (const float*)d_in[3 + 6 * i];
    const float* bp  = (const float*)d_in[4 + 6 * i];
    const int*   eg  = (const int*)d_in[5 + 6 * i];
    const int*   cl  = (const int*)d_in[6 + 6 * i];
    const int E = Earr[i];
    const int Ni = N[i], Nn = N[i + 1];
    const int* esrc = eg;
    const int* edst = eg + E;

    // 1) h = x @ W
    if (i == 0) {
      int tot = Ni * DIMC;
      gemm3<<<(tot + 255) / 256, 256, 0, stream>>>(x_in, Wp, Hbuf, Ni);
    } else {
      wt_convert<<<256, 256, 0, stream>>>(Wp, Wt);
      gemm_h<<<Ni / 16, 128, 0, stream>>>(Xbuf, Wt, Hbuf, Ni);   // Ni multiple of 16
    }

    // 2) attention dot products
    dots_kernel<<<(Ni + 3) / 4, 128, 0, stream>>>(Hbuf, asr, ads, hs, hd, Ni);

    // 3) init accumulators
    fill_u32<<<(Ni + 255) / 256, 256, 0, stream>>>(menc, ENC_NEG_INF, Ni);
    fill_u32<<<(Ni + 255) / 256, 256, 0, stream>>>((unsigned*)ssum, 0u, Ni);
    fill_u32<<<(Ni * DIMC + 255) / 256, 256, 0, stream>>>((unsigned*)Obuf, 0u, Ni * DIMC);
    fill_u32<<<2, 256, 0, stream>>>((unsigned*)stats, 0u, 2 * DIMC);

    // 4-6) edge softmax + aggregation
    edge_max<<<(E + 255) / 256, 256, 0, stream>>>(esrc, edst, hs, hd, menc, E);
    edge_exp<<<(E + 255) / 256, 256, 0, stream>>>(esrc, edst, hs, hd, menc, exbuf, ssum, E);
    edge_aggr<<<E, 256, 0, stream>>>(esrc, edst, exbuf, ssum, Hbuf, Obuf);

    // 7-8) +bias, relu, pool (Hbuf free after aggregation for i==2)
    float* poolDst = (i == 2) ? Hbuf : Xbuf;
    fill_u32<<<(Nn * DIMC + 255) / 256, 256, 0, stream>>>((unsigned*)poolDst, 0u, Nn * DIMC);
    pool_relu_max<<<Ni, 256, 0, stream>>>(Obuf, bp, cl, poolDst);

    // 9-11) feature normalization over all rows
    col_stats<<<128, 256, 0, stream>>>(poolDst, stats, Nn);
    col_finalize<<<1, 256, 0, stream>>>(stats, Nn);
    float* normDst = (i == 2) ? (float*)d_out : poolDst;
    col_apply<<<(Nn * DIMC + 255) / 256, 256, 0, stream>>>(poolDst, stats, normDst, Nn * DIMC);
  }
}
